// FeatureEmbedded_6356551598275
// MI455X (gfx1250) — compile-verified, matrix operand order checked by reading, not run on hardware
//
#include <hip/hip_runtime.h>

// ---------------- problem constants ----------------
#define Bz   128
#define Tz   512
#define Dz   192
#define Hz   512
#define Gz   1536   // 3*H
#define L1IN 1024   // 2*H

typedef __attribute__((ext_vector_type(16))) __bf16 v16bf;
typedef __attribute__((ext_vector_type(8)))  float  v8f;

union FragBF { v16bf v; uint4 u[2]; };
union Acc    { v8f v; float f[8]; };

__device__ __forceinline__ unsigned short f2bf(float f) {
  unsigned int u = __float_as_uint(f);
  u += 0x7FFFu + ((u >> 16) & 1u);           // round-to-nearest-even
  return (unsigned short)(u >> 16);
}
__device__ __forceinline__ float bf2f(unsigned short h) {
  return __uint_as_float(((unsigned int)h) << 16);
}

__device__ __forceinline__ v8f wmma_bf16(const FragBF& a, const FragBF& b, v8f c) {
  // D = A(16x32 bf16) * B(32x16 bf16) + C(16x16 f32)
  return __builtin_amdgcn_wmma_f32_16x16x32_bf16(false, a.v, false, b.v,
                                                 (short)0, c, false, false);
}

// ---------------- fp32 -> bf16 conversion ----------------
__global__ void cvt_f32_bf16(const float* __restrict__ in,
                             unsigned short* __restrict__ out, int n) {
  int i = blockIdx.x * blockDim.x + threadIdx.x;
  if (i < n) out[i] = f2bf(in[i]);
}

// ---------------- big time-parallel GEMM: C = A * W^T + bias ----------------
// A: [M,K] bf16 row-major, W: [N,K] bf16 row-major (PyTorch layout),
// bias: [N] f32, C: [M,N] bf16.
// Block: 256 threads = 8 waves; tile 64(M) x 128(N); wave computes 16x64.
__global__ __launch_bounds__(256)
void gemm_xw_bias(const unsigned short* __restrict__ A,
                  const unsigned short* __restrict__ W,
                  const float* __restrict__ bias,
                  unsigned short* __restrict__ C,
                  int M, int N, int K) {
  const int lane   = threadIdx.x & 31;
  const int w      = threadIdx.x >> 5;
  const int wm     = w & 3;
  const int wn     = w >> 2;
  const int lane15 = lane & 15;
  const int lhi    = (lane >> 4) << 3;       // 0 or 8
  const int m0     = blockIdx.y * 64 + wm * 16;
  const int n0     = blockIdx.x * 128 + wn * 64;

  Acc acc[4];
#pragma unroll
  for (int t = 0; t < 4; ++t)
#pragma unroll
    for (int r = 0; r < 8; ++r) acc[t].f[r] = 0.f;

  const unsigned short* arow = A + (size_t)(m0 + lane15) * K + lhi;
  for (int k0 = 0; k0 < K; k0 += 32) {
    FragBF a;
    a.u[0] = *reinterpret_cast<const uint4*>(arow + k0);
    a.u[1] = *reinterpret_cast<const uint4*>(arow + k0 + 16);
#pragma unroll
    for (int t = 0; t < 4; ++t) {
      const unsigned short* wr =
          W + (size_t)(n0 + t * 16 + lane15) * K + k0 + lhi;
      FragBF b;
      b.u[0] = *reinterpret_cast<const uint4*>(wr);
      b.u[1] = *reinterpret_cast<const uint4*>(wr + 16);
      acc[t].v = wmma_bf16(a, b, acc[t].v);
    }
  }

#pragma unroll
  for (int t = 0; t < 4; ++t) {
    const int col = n0 + t * 16 + lane15;
    const float bv = bias[col];
#pragma unroll
    for (int r = 0; r < 8; ++r) {
      const int row = m0 + lhi + r;          // C/D layout: M = r + 8*(lane>=16)
      C[(size_t)row * N + col] = f2bf(acc[t].f[r] + bv);
    }
  }
}

// ---------------- persistent bidirectional GRU scan ----------------
// grid.x = 16: dir = blk>>3, batch tile of 16 rows = (blk&7)*16.
// Per step: gh(16x1536) = h(16x512,bf16 in LDS) @ Whh^T via WMMA
//           (8 waves x 12 N-tiles), then fused gate update.
// gx tile for the *next* step is double-buffered into LDS with
// global_load_async_to_lds_b128 (ASYNCcnt), overlapping the WMMA phase.
__global__ __launch_bounds__(256)
void gru_scan(const unsigned short* __restrict__ gx,   // [2][B*T][G] bf16 (b_ih included)
              const unsigned short* __restrict__ Whh,  // [2][G][H] bf16
              const float* __restrict__ bhh,           // [2][G]
              const int* __restrict__ lengths,         // [B]
              unsigned short* __restrict__ h1out,      // layer0: [B*T][2H] bf16
              float* __restrict__ emb,                 // layer1: [B][2H] f32
              int layer) {
  extern __shared__ char smem[];
  float*          gh  = (float*)smem;                                       // 16*G f32
  float*          hf  = (float*)(smem + 16 * Gz * 4);                       // 16*H f32
  unsigned short* hb  = (unsigned short*)(smem + 16 * Gz * 4 + 16 * Hz * 4);// 16*H bf16
  int*            Ls  = (int*)(smem + 16 * Gz * 4 + 16 * Hz * 4 + 16 * Hz * 2);
  unsigned short* gxl = (unsigned short*)(smem + 16 * Gz * 4 + 16 * Hz * 4 +
                                          16 * Hz * 2 + 64);                // 2*16*G bf16

  const int dir    = blockIdx.x >> 3;
  const int b0     = (blockIdx.x & 7) * 16;
  const int tid    = threadIdx.x;
  const int lane   = tid & 31;
  const int w      = tid >> 5;
  const int lane15 = lane & 15;
  const int lhi    = (lane >> 4) << 3;
  const int c0     = w * 192;                // 12 N-tiles of 16 per wave

  for (int i = tid; i < 16 * Hz; i += 256) { hf[i] = 0.f; hb[i] = 0; }
  if (tid < 16) Ls[tid] = lengths[b0 + tid];
  __syncthreads();

  const unsigned short* Wd  = Whh + (size_t)dir * Gz * Hz;
  const float*          bd  = bhh + dir * Gz;
  const unsigned short* gxd = gx + (size_t)dir * (size_t)Bz * Tz * Gz;
  const unsigned gxlOff = (unsigned)(uintptr_t)gxl;   // LDS byte offset of gx buffers

  // Async-stage the 16x1536 bf16 gx tile for `step` into LDS buffer `bufsel`.
  // 3072 chunks of 16B; 256 threads x 12 async b128 copies (12 per wave -> ASYNCcnt+12).
  auto issue_gx = [&](int step, int bufsel) {
#pragma unroll
    for (int i = 0; i < 12; ++i) {
      const int c   = tid + 256 * i;         // 0..3071
      const int row = c / 192;               // 192 x 16B per row
      const int off = (c - row * 192) * 16;  // byte offset within row
      const int L   = Ls[row];
      const int tin = (dir == 0) ? step : ((step < L) ? (L - 1 - step) : step);
      const char* src = (const char*)gxd +
          (((size_t)(b0 + row) * Tz + tin) * Gz) * 2 + off;
      const unsigned dst = gxlOff + (unsigned)(bufsel * (16 * Gz * 2) +
                                               row * (Gz * 2) + off);
      asm volatile("global_load_async_to_lds_b128 %0, %1, off"
                   :: "v"(dst), "v"(src) : "memory");
    }
  };

  issue_gx(0, 0);                            // prefetch step 0

  for (int s = 0; s < Tz; ++s) {
    if (s + 1 < Tz) issue_gx(s + 1, (s + 1) & 1);   // overlaps phase-1 WMMAs

    // ---- phase 1: gh = h @ Whh^T ----
    Acc acc[12];
#pragma unroll
    for (int t = 0; t < 12; ++t)
#pragma unroll
      for (int r = 0; r < 8; ++r) acc[t].f[r] = 0.f;

    const unsigned short* hrow = hb + lane15 * Hz + lhi;
    for (int k0 = 0; k0 < Hz; k0 += 32) {
      FragBF a;                               // ds_load_b128 from LDS
      a.u[0] = *reinterpret_cast<const uint4*>(hrow + k0);
      a.u[1] = *reinterpret_cast<const uint4*>(hrow + k0 + 16);
#pragma unroll
      for (int t = 0; t < 12; ++t) {
        const unsigned short* wr =
            Wd + (size_t)(c0 + t * 16 + lane15) * Hz + k0 + lhi;
        FragBF b;
        b.u[0] = *reinterpret_cast<const uint4*>(wr);
        b.u[1] = *reinterpret_cast<const uint4*>(wr + 16);
        acc[t].v = wmma_bf16(a, b, acc[t].v);
      }
    }
#pragma unroll
    for (int t = 0; t < 12; ++t) {
      const int col = c0 + t * 16 + lane15;
#pragma unroll
      for (int r = 0; r < 8; ++r) gh[(lhi + r) * Gz + col] = acc[t].f[r];
    }

    // Retire this step's async gx batch (in-order per wave: <=12 outstanding
    // means the previous batch of 12 has landed); then make it block-visible.
    if (s + 1 < Tz) asm volatile("s_wait_asynccnt 0xc" ::: "memory");
    else            asm volatile("s_wait_asynccnt 0x0" ::: "memory");
    __syncthreads();

    // ---- phase 2: fused GRU gates (gx read from LDS) ----
    const unsigned short* gxs = gxl + (s & 1) * (16 * Gz);
    for (int u = tid; u < 16 * Hz; u += 256) {
      const int row = u >> 9;                 // /H
      const int j   = u & (Hz - 1);
      const int L   = Ls[row];
      const int tin = (dir == 0) ? s : ((s < L) ? (L - 1 - s) : s);
      const int b   = b0 + row;
      const float xr = bf2f(gxs[row * Gz + j]);
      const float xz = bf2f(gxs[row * Gz + Hz + j]);
      const float xn = bf2f(gxs[row * Gz + 2 * Hz + j]);
      const float hr = gh[row * Gz + j]          + bd[j];
      const float hz = gh[row * Gz + Hz + j]     + bd[Hz + j];
      const float hn = gh[row * Gz + 2 * Hz + j] + bd[2 * Hz + j];
      const float rg = 1.f / (1.f + __expf(-(xr + hr)));
      const float zg = 1.f / (1.f + __expf(-(xz + hz)));
      const float ng = tanhf(xn + rg * hn);
      const float hnew = (1.f - zg) * ng + zg * hf[row * Hz + j];
      hf[row * Hz + j] = hnew;
      hb[row * Hz + j] = f2bf(hnew);
      if (layer == 0) {
        // masked layer-0 output (reverse-scatter for backward dir)
        h1out[((size_t)b * Tz + tin) * (2 * Hz) + dir * Hz + j] =
            (tin < L) ? f2bf(hnew) : (unsigned short)0;
      } else {
        if (tin == L - 1) emb[(size_t)b * (2 * Hz) + dir * Hz + j] = hnew;
      }
    }
    __syncthreads();
  }
}

// ---------------- host launcher ----------------
extern "C" void kernel_launch(void* const* d_in, const int* in_sizes, int n_in,
                              void* d_out, int out_size, void* d_ws, size_t ws_size,
                              hipStream_t stream) {
  const float* x        = (const float*)d_in[0];
  const int*   lengths  = (const int*)d_in[1];
  const float* W_ih_l0  = (const float*)d_in[2];
  const float* W_hh_l0  = (const float*)d_in[3];
  const float* b_ih_l0  = (const float*)d_in[4];
  const float* b_hh_l0  = (const float*)d_in[5];
  const float* W_ih_l1  = (const float*)d_in[6];
  const float* W_hh_l1  = (const float*)d_in[7];
  const float* b_ih_l1  = (const float*)d_in[8];
  const float* b_hh_l1  = (const float*)d_in[9];
  float* out = (float*)d_out;

  char* ws = (char*)d_ws;
  size_t off = 0;
  auto alloc = [&](size_t bytes) -> void* {
    void* p = ws + off;
    off += (bytes + 255) & ~(size_t)255;
    return p;
  };
  const size_t M = (size_t)Bz * Tz;   // 65536 rows
  unsigned short* xbf  = (unsigned short*)alloc(M * Dz * 2);
  unsigned short* h1bf = (unsigned short*)alloc(M * L1IN * 2);
  unsigned short* gxb  = (unsigned short*)alloc(2 * M * Gz * 2);
  unsigned short* wih0 = (unsigned short*)alloc((size_t)2 * Gz * Dz * 2);
  unsigned short* whh0 = (unsigned short*)alloc((size_t)2 * Gz * Hz * 2);
  unsigned short* wih1 = (unsigned short*)alloc((size_t)2 * Gz * L1IN * 2);
  unsigned short* whh1 = (unsigned short*)alloc((size_t)2 * Gz * Hz * 2);

  auto cvt = [&](const float* src, unsigned short* dst, int n) {
    cvt_f32_bf16<<<(n + 255) / 256, 256, 0, stream>>>(src, dst, n);
  };
  cvt(x,       xbf,  (int)(M * Dz));
  cvt(W_ih_l0, wih0, 2 * Gz * Dz);
  cvt(W_hh_l0, whh0, 2 * Gz * Hz);
  cvt(W_ih_l1, wih1, 2 * Gz * L1IN);
  cvt(W_hh_l1, whh1, 2 * Gz * Hz);

  const int smemScan = 16 * Gz * 4        // gh  (96 KB)
                     + 16 * Hz * 4        // hf  (32 KB)
                     + 16 * Hz * 2        // hb  (16 KB)
                     + 64                 // Ls
                     + 2 * 16 * Gz * 2;   // gx double buffer (96 KB)
  hipFuncSetAttribute((const void*)gru_scan,
                      hipFuncAttributeMaxDynamicSharedMemorySize, smemScan);

  dim3 blk(256);
  // ---- layer 0 ----
  {
    dim3 grd(Gz / 128, (unsigned)(M / 64));
    for (int d = 0; d < 2; ++d)
      gemm_xw_bias<<<grd, blk, 0, stream>>>(
          xbf, wih0 + (size_t)d * Gz * Dz, b_ih_l0 + (size_t)d * Gz,
          gxb + (size_t)d * M * Gz, (int)M, Gz, Dz);
    gru_scan<<<16, blk, smemScan, stream>>>(gxb, whh0, b_hh_l0, lengths,
                                            h1bf, nullptr, 0);
  }
  // ---- layer 1 ----
  {
    dim3 grd(Gz / 128, (unsigned)(M / 64));
    for (int d = 0; d < 2; ++d)
      gemm_xw_bias<<<grd, blk, 0, stream>>>(
          h1bf, wih1 + (size_t)d * Gz * L1IN, b_ih_l1 + (size_t)d * Gz,
          gxb + (size_t)d * M * Gz, (int)M, Gz, L1IN);
    gru_scan<<<16, blk, smemScan, stream>>>(gxb, whh1, b_hh_l1, lengths,
                                            nullptr, out, 1);
  }
}
